// CrossModalAttention2_90847148245447
// MI455X (gfx1250) — matrix operand order repaired and checked by reading, not compiled, not run on hardware
//
#include <hip/hip_runtime.h>

// ---------------------------------------------------------------------------
// CDNA5 (gfx1250) cross-modal attention, bf16 WMMA pipeline.
// Sizes: B=8, HW=4096, NL=256, DIM=256, HEADS=8, dk=dv=32.
// Workspace requirement: ~206 MB (see layout in kernel_launch).
// ---------------------------------------------------------------------------

typedef __bf16 bf16_t;
typedef __attribute__((ext_vector_type(16))) __bf16 v16bf;
typedef __attribute__((ext_vector_type(8)))  __bf16 v8bf;
typedef __attribute__((ext_vector_type(8)))  float  v8f;

static __device__ __forceinline__ bf16_t f2bf(float f) {
  unsigned u = __builtin_bit_cast(unsigned, f);
  unsigned r = (u + 0x7FFFu + ((u >> 16) & 1u)) >> 16;   // round-to-nearest-even
  unsigned short h = (unsigned short)r;
  return __builtin_bit_cast(bf16_t, h);
}
static __device__ __forceinline__ float bf2f(bf16_t b) {
  unsigned short h = __builtin_bit_cast(unsigned short, b);
  unsigned u = ((unsigned)h) << 16;
  return __builtin_bit_cast(float, u);
}

static __device__ __forceinline__ v8f wmma_bf16(v16bf a, v16bf b, v8f c) {
  // D = A(16x32 bf16) * B(32x16 bf16) + C(16x16 f32)
  return __builtin_amdgcn_wmma_f32_16x16x32_bf16(false, a, false, b, (short)0, c,
                                                 false, false);
}

// A tile 16x32 (MxK), row-major source with leading dim lda.
// Contiguous-K layout used here (B uses the same): lane(0..15)=row M, half=lane>>4; elems e -> K=half*16+e,
// i.e. 16 contiguous K values per lane (two 16-byte loads).
static __device__ __forceinline__ v16bf load_a_tile(const bf16_t* __restrict__ A,
                                                    int lda, int row0, int k0,
                                                    int lane, int M) {
  int r = row0 + (lane & 15);
  if (r >= M) r = M - 1;                       // clamp (store-guard makes it harmless)
  int half = lane >> 4;
  const bf16_t* p = A + (size_t)r * lda + k0 + half * 16;
  v8bf lo = *(const v8bf*)(p);
  v8bf hi = *(const v8bf*)(p + 8);
  v16bf out;
#pragma unroll
  for (int i = 0; i < 8; ++i) { out[i] = lo[i]; out[8 + i] = hi[i]; }
  return out;
}

// B tile 32x16 (KxN) for C = A*B^T. Source is row-major [N,K] (ldb).
// Contiguous-K layout used here (matches load_a_tile): lane(0..15)=col N, khalf=lane>>4; elems e -> K=khalf*16+e,
// i.e. 16 contiguous K values per lane (two 16-byte loads).
static __device__ __forceinline__ v16bf load_b_tile(const bf16_t* __restrict__ Bm,
                                                    int ldb, int col0, int k0,
                                                    int lane, int N) {
  int n = col0 + (lane & 15);
  if (n >= N) n = N - 1;
  int kh = lane >> 4;
  const bf16_t* p = Bm + (size_t)n * ldb + k0 + kh * 16;
  v8bf lo = *(const v8bf*)(p);
  v8bf hi = *(const v8bf*)(p + 8);
  v16bf out;
#pragma unroll
  for (int i = 0; i < 8; ++i) { out[i] = lo[i]; out[8 + i] = hi[i]; }
  return out;
}

// C/D 16x16 f32 layout: lane N = lane&15; VGPR i -> M = i + (lane>>4)*8.
static __device__ __forceinline__ void store_tile(float* C, bf16_t* Cb,
                                                  const float* __restrict__ bias,
                                                  int ldc, int row0, int col0,
                                                  int M, int N, v8f d,
                                                  float alpha, int lane) {
  int n = col0 + (lane & 15);
  if (n >= N) return;
  int rbase = row0 + (lane >> 4) * 8;
  float badd = bias ? bias[n] : 0.0f;
#pragma unroll
  for (int i = 0; i < 8; ++i) {
    int m = rbase + i;
    if (m < M) {
      float v = d[i] * alpha + badd;
      if (C)  C[(size_t)m * ldc + n] = v;
      if (Cb) Cb[(size_t)m * ldc + n] = f2bf(v);
    }
  }
}

// Generic batched GEMM: C[z] = alpha * A[z] * B[z]^T (+bias), bf16 in, f32 acc.
// z decomposes as (b = z/zdiv, h = z%zdiv); per-operand (b,h) element strides.
// Block = 256 threads = 8 waves (4 along M x 2 along N); wave tile 32x32 =
// 2x2 WMMA accumulators. Block tile 128(M) x 64(N). K must be a multiple of 32.
__global__ void __launch_bounds__(256)
gemm_bf16_wmma_kernel(const bf16_t* __restrict__ A, const bf16_t* __restrict__ Bm,
                      float* C, bf16_t* Cb, const float* __restrict__ bias,
                      int M, int N, int K, int lda, int ldb, int ldc, float alpha,
                      int zdiv,
                      long long sAb, long long sAh,
                      long long sBb, long long sBh,
                      long long sCb, long long sCh) {
  int z  = blockIdx.z;
  int bb = z / zdiv, hh = z % zdiv;
  A  += (size_t)(bb * sAb + hh * sAh);
  Bm += (size_t)(bb * sBb + hh * sBh);
  long long coff = bb * sCb + hh * sCh;
  if (C)  C  += (size_t)coff;
  if (Cb) Cb += (size_t)coff;

  int wave = threadIdx.x >> 5;
  int lane = threadIdx.x & 31;
  int wm = wave & 3;                 // 4 waves along M
  int wn = wave >> 2;                // 2 waves along N
  int row0 = blockIdx.y * 128 + wm * 32;
  int col0 = blockIdx.x * 64 + wn * 32;

  v8f acc00 = {}, acc01 = {}, acc10 = {}, acc11 = {};
  for (int k0 = 0; k0 < K; k0 += 32) {
    if (k0 + 32 < K) {   // prefetch next K-step (global_prefetch_b8)
      __builtin_prefetch(A + (size_t)(row0 + (lane & 15)) * lda + k0 + 32, 0, 1);
      __builtin_prefetch(Bm + (size_t)(col0 + (lane & 15)) * ldb + k0 + 32, 0, 1);
    }
    v16bf a0 = load_a_tile(A, lda, row0,      k0, lane, M);
    v16bf a1 = load_a_tile(A, lda, row0 + 16, k0, lane, M);
    v16bf b0 = load_b_tile(Bm, ldb, col0,      k0, lane, N);
    v16bf b1 = load_b_tile(Bm, ldb, col0 + 16, k0, lane, N);
    acc00 = wmma_bf16(a0, b0, acc00);
    acc01 = wmma_bf16(a0, b1, acc01);
    acc10 = wmma_bf16(a1, b0, acc10);
    acc11 = wmma_bf16(a1, b1, acc11);
  }
  store_tile(C, Cb, bias, ldc, row0,      col0,      M, N, acc00, alpha, lane);
  store_tile(C, Cb, bias, ldc, row0,      col0 + 16, M, N, acc01, alpha, lane);
  store_tile(C, Cb, bias, ldc, row0 + 16, col0,      M, N, acc10, alpha, lane);
  store_tile(C, Cb, bias, ldc, row0 + 16, col0 + 16, M, N, acc11, alpha, lane);
}

// f32 -> bf16 elementwise.
__global__ void cvt_f32_bf16_kernel(const float* __restrict__ in,
                                    bf16_t* __restrict__ out, size_t n) {
  size_t i = (size_t)blockIdx.x * blockDim.x + threadIdx.x;
  size_t stride = (size_t)gridDim.x * blockDim.x;
  for (; i < n; i += stride) out[i] = f2bf(in[i]);
}

// vT[b, c, l] = v[b, l, c]  (B=8, 256x256 per batch, bf16)
__global__ void transpose_v_kernel(const bf16_t* __restrict__ v,
                                   bf16_t* __restrict__ vT) {
  int i = blockIdx.x * blockDim.x + threadIdx.x;
  if (i >= 8 * 256 * 256) return;
  int b = i >> 16, c = (i >> 8) & 255, l = i & 255;
  vT[((size_t)b << 16) + (c << 8) + l] = v[((size_t)b << 16) + (l << 8) + c];
}

// InstanceNorm over L rows per (batch, channel). X: [gridDim.x, L, C=256] f32.
// 1024 threads = 256 channels x 4 row-groups; fully coalesced.
__global__ void __launch_bounds__(1024)
instnorm_kernel(const float* __restrict__ X, float* outF, bf16_t* outB,
                int L, int C) {
  const float* x = X + (size_t)blockIdx.x * L * C;
  int col = threadIdx.x & 255;
  int rg  = threadIdx.x >> 8;
  int rows = L >> 2;
  int q0 = rg * rows, q1 = q0 + rows;
  float s = 0.f, s2 = 0.f;
  for (int q = q0; q < q1; ++q) {
    float v = x[(size_t)q * C + col];
    s += v; s2 += v * v;
  }
  __shared__ float sh[4][256], sh2[4][256];
  __shared__ float smu[256], siv[256];
  sh[rg][col] = s; sh2[rg][col] = s2;
  __syncthreads();
  if (rg == 0) {
    float ts  = sh[0][col] + sh[1][col] + sh[2][col] + sh[3][col];
    float ts2 = sh2[0][col] + sh2[1][col] + sh2[2][col] + sh2[3][col];
    float mu  = ts / (float)L;
    float var = ts2 / (float)L - mu * mu;
    smu[col] = mu;
    siv[col] = rsqrtf(var + 1e-5f);
  }
  __syncthreads();
  float mu = smu[col], iv = siv[col];
  float*  of = outF ? outF + (size_t)blockIdx.x * L * C : nullptr;
  bf16_t* ob = outB ? outB + (size_t)blockIdx.x * L * C : nullptr;
  for (int q = q0; q < q1; ++q) {
    size_t idx = (size_t)q * C + col;
    float v = (x[idx] - mu) * iv;
    if (of) of[idx] = v;
    if (ob) ob[idx] = f2bf(v);
  }
}

// In-place column softmax over the HW=4096 axis. S: [64 slabs][4096][256] bf16.
// One block per (b,h): 1024 threads = 256 columns x 4 row-groups, online
// max/rescaled-sum, LDS combine, then rewrite P = exp(x-m)/s in place.
// Safe in place: each thread exclusively owns its 1024-row segment of one column.
__global__ void __launch_bounds__(1024)
colsoftmax_kernel(bf16_t* S) {
  const size_t slab = (size_t)4096 * 256;
  bf16_t* s = S + (size_t)blockIdx.x * slab;
  int col = threadIdx.x & 255;
  int rg  = threadIdx.x >> 8;
  int q0 = rg * 1024, q1 = q0 + 1024;
  float m = -3.0e38f, sum = 0.f;
  for (int q = q0; q < q1; ++q) {
    float x = bf2f(s[(size_t)q * 256 + col]);
    if (x > m) { sum *= __expf(m - x); m = x; }
    sum += __expf(x - m);
  }
  __shared__ float shm[4][256], shs[4][256];
  __shared__ float fm[256], fr[256];
  shm[rg][col] = m; shs[rg][col] = sum;
  __syncthreads();
  if (rg == 0) {
    float mm = fmaxf(fmaxf(shm[0][col], shm[1][col]),
                     fmaxf(shm[2][col], shm[3][col]));
    float ss = shs[0][col] * __expf(shm[0][col] - mm)
             + shs[1][col] * __expf(shm[1][col] - mm)
             + shs[2][col] * __expf(shm[2][col] - mm)
             + shs[3][col] * __expf(shm[3][col] - mm);
    fm[col] = mm;
    fr[col] = 1.0f / ss;
  }
  __syncthreads();
  float mm = fm[col], rr = fr[col];
  for (int q = q0; q < q1; ++q) {
    size_t idx = (size_t)q * 256 + col;
    s[idx] = f2bf(__expf(bf2f(s[idx]) - mm) * rr);
  }
}

extern "C" void kernel_launch(void* const* d_in, const int* in_sizes, int n_in,
                              void* d_out, int out_size, void* d_ws, size_t ws_size,
                              hipStream_t stream) {
  const float* visual = (const float*)d_in[0];
  const float* seg    = (const float*)d_in[1];
  const float* Wq = (const float*)d_in[2];
  const float* bq = (const float*)d_in[3];
  const float* Wk = (const float*)d_in[4];
  const float* bk = (const float*)d_in[5];
  const float* Wv = (const float*)d_in[6];
  const float* bv = (const float*)d_in[7];
  const float* Wo = (const float*)d_in[8];
  const float* bo = (const float*)d_in[9];
  float* out = (float*)d_out;

  const int B = 8, HW = 4096, NL = 256, DIM = 256, H = 8;
  const long long SLAB = (long long)HW * DIM;   // 4096*256 elements

  // ---- workspace layout (256B-aligned regions; ~206 MB total) ----
  char* ws = (char*)d_ws;
  size_t off = 0;
  auto take = [&](size_t bytes) -> char* {
    char* p = ws + off;
    off += (bytes + 255) & ~(size_t)255;
    return p;
  };
  bf16_t* visbf  = (bf16_t*)take((size_t)B * HW * DIM * 2);   // 16 MB
  bf16_t* segbf  = (bf16_t*)take((size_t)B * NL * DIM * 2);   //  1 MB
  bf16_t* wqb    = (bf16_t*)take((size_t)DIM * DIM * 2);
  bf16_t* wkb    = (bf16_t*)take((size_t)DIM * DIM * 2);
  bf16_t* wvb    = (bf16_t*)take((size_t)DIM * DIM * 2);
  bf16_t* wob    = (bf16_t*)take((size_t)DIM * DIM * 2);
  float*  qbuf   = (float*) take((size_t)B * HW * DIM * 4);   // 32 MB (reused: oproj)
  bf16_t* qnorm  = (bf16_t*)take((size_t)B * HW * DIM * 2);   // 16 MB
  bf16_t* kbf    = (bf16_t*)take((size_t)B * NL * DIM * 2);   //  1 MB
  bf16_t* vbf    = (bf16_t*)take((size_t)B * NL * DIM * 2);   //  1 MB
  bf16_t* vT     = (bf16_t*)take((size_t)B * NL * DIM * 2);   //  1 MB
  bf16_t* simbuf = (bf16_t*)take((size_t)B * H * HW * NL * 2);// 128 MB (sim -> P in place)
  bf16_t* attnbf = visbf;  // alias: visual bf16 is dead after q projection

  // ---- 1) f32 -> bf16 conversions ----
  cvt_f32_bf16_kernel<<<2048, 256, 0, stream>>>(visual, visbf, (size_t)B * HW * DIM);
  cvt_f32_bf16_kernel<<<512,  256, 0, stream>>>(seg,    segbf, (size_t)B * NL * DIM);
  cvt_f32_bf16_kernel<<<64,   256, 0, stream>>>(Wq, wqb, (size_t)DIM * DIM);
  cvt_f32_bf16_kernel<<<64,   256, 0, stream>>>(Wk, wkb, (size_t)DIM * DIM);
  cvt_f32_bf16_kernel<<<64,   256, 0, stream>>>(Wv, wvb, (size_t)DIM * DIM);
  cvt_f32_bf16_kernel<<<64,   256, 0, stream>>>(Wo, wob, (size_t)DIM * DIM);

  // ---- 2) q = visual @ Wq^T + bq  (f32 out, then InstanceNorm -> bf16) ----
  gemm_bf16_wmma_kernel<<<dim3(4, 256, 1), 256, 0, stream>>>(
      visbf, wqb, qbuf, nullptr, bq,
      B * HW, DIM, DIM, DIM, DIM, DIM, 1.0f, H, 0, 0, 0, 0, 0, 0);

  // ---- 3) k,v = seg @ Wk^T/Wv^T + b  (bf16 out) ----
  gemm_bf16_wmma_kernel<<<dim3(4, 16, 1), 256, 0, stream>>>(
      segbf, wkb, nullptr, kbf, bk,
      B * NL, DIM, DIM, DIM, DIM, DIM, 1.0f, H, 0, 0, 0, 0, 0, 0);
  gemm_bf16_wmma_kernel<<<dim3(4, 16, 1), 256, 0, stream>>>(
      segbf, wvb, nullptr, vbf, bv,
      B * NL, DIM, DIM, DIM, DIM, DIM, 1.0f, H, 0, 0, 0, 0, 0, 0);
  transpose_v_kernel<<<2048, 256, 0, stream>>>(vbf, vT);

  // ---- 4) InstanceNorm(q) over HW per (b,c) -> qnorm bf16 ----
  instnorm_kernel<<<B, 1024, 0, stream>>>(qbuf, nullptr, qnorm, HW, DIM);

  // ---- 5) sim[b,h] = (qh @ kh^T) / 16  -> bf16, per-(b,h) slabs ----
  gemm_bf16_wmma_kernel<<<dim3(4, 32, B * H), 256, 0, stream>>>(
      qnorm, kbf, nullptr, simbuf, nullptr,
      HW, NL, 32, DIM, DIM, NL, 0.0625f, H,
      /*A:*/ (long long)HW * DIM, 32,
      /*B:*/ (long long)NL * DIM, 32,
      /*C:*/ (long long)H * SLAB, SLAB);

  // ---- 6) softmax over the query (HW) axis, in place ----
  colsoftmax_kernel<<<B * H, 1024, 0, stream>>>(simbuf);

  // ---- 7) attn[b,q,h*32+d] = P[b,h] @ vh[b,h]  (bf16 out) ----
  gemm_bf16_wmma_kernel<<<dim3(1, 32, B * H), 256, 0, stream>>>(
      simbuf, vT, nullptr, attnbf, nullptr,
      HW, 32, NL, NL, NL, DIM, 1.0f, H,
      /*A:*/ (long long)H * SLAB, SLAB,
      /*B:*/ (long long)NL * DIM, (long long)32 * NL,
      /*C:*/ (long long)HW * DIM, 32);

  // ---- 8) o = attn @ Wo^T + bo (f32), then InstanceNorm -> d_out ----
  gemm_bf16_wmma_kernel<<<dim3(4, 256, 1), 256, 0, stream>>>(
      attnbf, wob, qbuf, nullptr, bo,
      B * HW, DIM, DIM, DIM, DIM, DIM, 1.0f, H, 0, 0, 0, 0, 0, 0);
  instnorm_kernel<<<B, 1024, 0, stream>>>(qbuf, out, nullptr, HW, DIM);
}
